// MMDLoss_7997229105699
// MI455X (gfx1250) — matrix operand order repaired and checked by reading, not compiled, not run on hardware
//
#include <hip/hip_runtime.h>

// ---------------------------------------------------------------------------
// MMD loss for X,Y in R^{4096x512} fp32.
//   Pass 1 (cheap): row norms sq[i], S = sum sq, colsum m[d]  ->  analytic
//                   bandwidth bw = (2nS - 2|m|^2)/(n^2-n).
//   Pass 2 (hot):   tiled Gram Z Z^T via bf16 hi/lo split WMMA (3x
//                   v_wmma_f32_16x16x32_bf16 per 16x16x32 step, f32 acc),
//                   fused epilogue: D2 -> e=exp(-D2/(4bw)),
//                   K = e + e^2 + e^4 + e^8 + e^16, quadrant reduction.
// ---------------------------------------------------------------------------

typedef __bf16 bf16_t;
typedef __bf16 bf16x8  __attribute__((ext_vector_type(8)));
typedef __bf16 bf16x16 __attribute__((ext_vector_type(16)));
typedef float  f32x4   __attribute__((ext_vector_type(4)));
typedef float  f32x8   __attribute__((ext_vector_type(8)));

#define NROWS 8192          // n = 2*4096
#define NX    4096
#define DDIM  512
#define BT    128           // block tile (output)
#define KT    32            // k tile (bf16 WMMA K)
#define LDSK  40            // padded k-stride in bf16 elements (avoids bank conflicts)

// workspace layout (float indices)
#define WS_S    0           // sum of row norms
#define WS_MSQ  1           // |colsum|^2
#define WS_SLOT 2           // [2..4]: XX, XY(+YX), YY partial sums
#define WS_COL  8           // [8..520): column sums
#define WS_SQ   1024        // [1024..1024+8192): row squared norms

__device__ __forceinline__ const float* zrow(const float* X, const float* Y, int r) {
    return (r < NX) ? (X + (size_t)r * DDIM) : (Y + (size_t)(r - NX) * DDIM);
}

__device__ __forceinline__ bf16x16 load_frag(const bf16_t* p0, const bf16_t* p1) {
    bf16x8 c0 = *reinterpret_cast<const bf16x8*>(p0);   // 16B LDS load
    bf16x8 c1 = *reinterpret_cast<const bf16x8*>(p1);   // 16B LDS load
    return __builtin_shufflevector(c0, c1, 0,1,2,3,4,5,6,7,8,9,10,11,12,13,14,15);
}

// ---------------- pass-1 kernels ----------------

__global__ void init_kernel(float* ws) {
    for (int i = threadIdx.x; i < 1024; i += 256) ws[i] = 0.0f;
}

__global__ __launch_bounds__(256) void rowsq_kernel(const float* X, const float* Y, float* ws) {
    const int lane = threadIdx.x & 31;
    const int wv   = threadIdx.x >> 5;              // wave32: 8 waves / block
    const int row  = blockIdx.x * 8 + wv;
    const float* zr = zrow(X, Y, row);
    float s = 0.0f;
    #pragma unroll
    for (int d = lane; d < DDIM; d += 32) { float v = zr[d]; s += v * v; }
    #pragma unroll
    for (int m = 16; m > 0; m >>= 1) s += __shfl_xor(s, m, 32);
    if (lane == 0) { ws[WS_SQ + row] = s; atomicAdd(&ws[WS_S], s); }
}

__global__ __launch_bounds__(256) void colsum_kernel(const float* X, const float* Y, float* ws) {
    const int t  = threadIdx.x;
    const int r0 = blockIdx.x * 128;
    float s0 = 0.0f, s1 = 0.0f;
    for (int r = 0; r < 128; ++r) {
        const float* zr = zrow(X, Y, r0 + r);       // coalesced across t
        s0 += zr[t];
        s1 += zr[t + 256];
    }
    atomicAdd(&ws[WS_COL + t], s0);
    atomicAdd(&ws[WS_COL + t + 256], s1);
}

__global__ void msq_kernel(float* ws) {
    __shared__ float red[256];
    const int t = threadIdx.x;
    float a = ws[WS_COL + t], b = ws[WS_COL + t + 256];
    red[t] = a * a + b * b;
    __syncthreads();
    for (int s = 128; s > 0; s >>= 1) { if (t < s) red[t] += red[t + s]; __syncthreads(); }
    if (t == 0) ws[WS_MSQ] = red[0];
}

// ---------------- pass-2: Gram + RBF + quadrant reduction ----------------

__global__ __launch_bounds__(256) void mmd_kernel(const float* __restrict__ X,
                                                  const float* __restrict__ Y,
                                                  float* __restrict__ ws) {
    __shared__ bf16_t Ahi[BT * LDSK], Alo[BT * LDSK];
    __shared__ bf16_t Bhi[BT * LDSK], Blo[BT * LDSK];
    __shared__ float  red[256];

    const int i0   = blockIdx.y * BT;
    const int j0   = blockIdx.x * BT;
    const int tid  = threadIdx.x;
    const int lane = tid & 31;
    const int wv   = tid >> 5;                      // 8 waves, wave wv owns rows [wv*16, wv*16+16)

    // analytic bandwidth from pass-1 stats
    const double n  = (double)NROWS;
    const double bw = (2.0 * n * (double)ws[WS_S] - 2.0 * (double)ws[WS_MSQ]) / (n * n - n);
    const float  ninvbw4 = (float)(-0.25 / bw);     // e = exp(-D2/(4 bw))

    f32x8 acc[8];
    #pragma unroll
    for (int t = 0; t < 8; ++t)
        #pragma unroll
        for (int r = 0; r < 8; ++r) acc[t][r] = 0.0f;

    // tile staging: 2 threads per row, 16 floats (64B) each
    const int lr = tid >> 1;                        // 0..127 row within tile
    const int lh = (tid & 1) * 16;                  // k offset 0 or 16
    const float* arow = zrow(X, Y, i0 + lr);
    const float* brow = zrow(X, Y, j0 + lr);

    for (int kt = 0; kt < DDIM; kt += KT) {
        // ---- global fp32 loads (L2-resident working set) ----
        f32x4 av[4], bv[4];
        #pragma unroll
        for (int c = 0; c < 4; ++c) {
            av[c] = *reinterpret_cast<const f32x4*>(arow + kt + lh + 4 * c);
            bv[c] = *reinterpret_cast<const f32x4*>(brow + kt + lh + 4 * c);
        }
        __syncthreads();                            // previous iter's LDS reads done
        // ---- fp32 -> bf16 hi/lo split, 16B LDS stores ----
        {
            float xa[16], xb[16];
            #pragma unroll
            for (int c = 0; c < 4; ++c)
                #pragma unroll
                for (int u = 0; u < 4; ++u) { xa[4*c+u] = av[c][u]; xb[4*c+u] = bv[c][u]; }
            bf16x8 ah0, ah1, al0, al1, bh0, bh1, bl0, bl1;
            #pragma unroll
            for (int u = 0; u < 8; ++u) {
                float x = xa[u];     bf16_t h = (bf16_t)x; ah0[u] = h; al0[u] = (bf16_t)(x - (float)h);
                x = xa[u + 8];       h = (bf16_t)x;        ah1[u] = h; al1[u] = (bf16_t)(x - (float)h);
                x = xb[u];           h = (bf16_t)x;        bh0[u] = h; bl0[u] = (bf16_t)(x - (float)h);
                x = xb[u + 8];       h = (bf16_t)x;        bh1[u] = h; bl1[u] = (bf16_t)(x - (float)h);
            }
            const int base = lr * LDSK + lh;
            *reinterpret_cast<bf16x8*>(&Ahi[base])     = ah0;
            *reinterpret_cast<bf16x8*>(&Ahi[base + 8]) = ah1;
            *reinterpret_cast<bf16x8*>(&Alo[base])     = al0;
            *reinterpret_cast<bf16x8*>(&Alo[base + 8]) = al1;
            *reinterpret_cast<bf16x8*>(&Bhi[base])     = bh0;
            *reinterpret_cast<bf16x8*>(&Bhi[base + 8]) = bh1;
            *reinterpret_cast<bf16x8*>(&Blo[base])     = bl0;
            *reinterpret_cast<bf16x8*>(&Blo[base + 8]) = bl1;
        }
        __syncthreads();

        // ---- A fragment (16-bit 16x32 layout, §7.12.2): lane group g=lane>>4
        //      holds K = [8g,8g+8) in v0-3 and [8g+16,8g+24) in v4-7, M = lane&15
        const int rA  = wv * 16 + (lane & 15);
        const int kbA = (lane >> 4) * 8;
        const bf16x16 afhi = load_frag(&Ahi[rA * LDSK + kbA], &Ahi[rA * LDSK + kbA + 16]);
        const bf16x16 aflo = load_frag(&Alo[rA * LDSK + kbA], &Alo[rA * LDSK + kbA + 16]);

        // ---- B fragments: N = lane&15 (row of Zj), same K split as A
        const int kbB = (lane >> 4) * 8;
        #pragma unroll
        for (int t = 0; t < 8; ++t) {
            const int rB = t * 16 + (lane & 15);
            const bf16x16 bfhi = load_frag(&Bhi[rB * LDSK + kbB], &Bhi[rB * LDSK + kbB + 16]);
            const bf16x16 bflo = load_frag(&Blo[rB * LDSK + kbB], &Blo[rB * LDSK + kbB + 16]);
            // dot(z_i, z_j) ~= hi.hi + hi.lo + lo.hi  (f32 accumulate)
            acc[t] = __builtin_amdgcn_wmma_f32_16x16x32_bf16(false, afhi, false, bfhi,
                                                             (short)0, acc[t], false, false);
            acc[t] = __builtin_amdgcn_wmma_f32_16x16x32_bf16(false, afhi, false, bflo,
                                                             (short)0, acc[t], false, false);
            acc[t] = __builtin_amdgcn_wmma_f32_16x16x32_bf16(false, aflo, false, bfhi,
                                                             (short)0, acc[t], false, false);
        }
    }

    // ---- fused RBF epilogue; C layout: VGPR r, lane l -> M = r + 8*(l>>4), N = l&15
    const int g   = lane >> 4;
    const int nlo = lane & 15;
    float sqi[8];
    #pragma unroll
    for (int r = 0; r < 8; ++r) sqi[r] = ws[WS_SQ + i0 + wv * 16 + r + 8 * g];

    float local = 0.0f;
    #pragma unroll
    for (int t = 0; t < 8; ++t) {
        const float sqj = ws[WS_SQ + j0 + t * 16 + nlo];
        #pragma unroll
        for (int r = 0; r < 8; ++r) {
            float d2 = fmaxf(sqi[r] + sqj - 2.0f * acc[t][r], 0.0f);
            float e  = __expf(d2 * ninvbw4);        // exp(-D2/(4bw))
            float e2 = e * e, e4 = e2 * e2, e8 = e4 * e4, e16 = e8 * e8;
            local += e + e2 + e4 + e8 + e16;        // 5 bandwidths, powers of 2
        }
    }

    red[tid] = local;
    __syncthreads();
    for (int s = 128; s > 0; s >>= 1) { if (tid < s) red[tid] += red[tid + s]; __syncthreads(); }
    if (tid == 0) {
        const int slot = (i0 < NX) ? ((j0 < NX) ? 0 : 1) : ((j0 < NX) ? 1 : 2);
        atomicAdd(&ws[WS_SLOT + slot], red[0]);
    }
}

__global__ void finalize_kernel(const float* ws, float* out) {
    if (threadIdx.x == 0) {
        const double denom = (double)NX * (double)NX;
        // XX - 2*XY + YY ; slot1 holds XY+YX = 2*XY_sum
        out[0] = (float)(((double)ws[WS_SLOT + 0] - (double)ws[WS_SLOT + 1]
                          + (double)ws[WS_SLOT + 2]) / denom);
    }
}

// ---------------------------------------------------------------------------

extern "C" void kernel_launch(void* const* d_in, const int* in_sizes, int n_in,
                              void* d_out, int out_size, void* d_ws, size_t ws_size,
                              hipStream_t stream) {
    const float* X = (const float*)d_in[0];
    const float* Y = (const float*)d_in[1];
    float* out = (float*)d_out;
    float* ws  = (float*)d_ws;   // needs (1024 + 8192) floats = 36 KB
    (void)in_sizes; (void)n_in; (void)out_size; (void)ws_size;

    init_kernel<<<1, 256, 0, stream>>>(ws);
    rowsq_kernel<<<NROWS / 8, 256, 0, stream>>>(X, Y, ws);
    colsum_kernel<<<NROWS / 128, 256, 0, stream>>>(X, Y, ws);
    msq_kernel<<<1, 256, 0, stream>>>(ws);

    dim3 grid(NROWS / BT, NROWS / BT);               // 64 x 64 = 4096 blocks
    mmd_kernel<<<grid, 256, 0, stream>>>(X, Y, ws);

    finalize_kernel<<<1, 32, 0, stream>>>(ws, out);
}
